// SimilarityMatrixLayer_59923383714505
// MI455X (gfx1250) — matrix operand order repaired and checked by reading, not compiled, not run on hardware
//
#include <hip/hip_runtime.h>
#include <hip/hip_bf16.h>
#include <math.h>

// ---------------------------------------------------------------------------
// SimilarityMatrixLayer for MI455X (gfx1250, wave32, WMMA)
//   e = inputs.reshape(2048,16,512); cmean = mean_m(e)
//   norm = sqrt(max(sum(cmean^2), 1e-6))
//   S = e @ cmeanT / norm ; out = clip(|w|*S + b, 0, 1)
// Round 3: SADDR-form async-to-LDS (32-bit per-lane offsets, uniform SGPR
// base) to kill v_add_nc_u64 chains; non-temporal output stores so the
// 256MB write stream doesn't evict the L2-resident bf16 operands.
// ---------------------------------------------------------------------------

typedef __bf16 bf16_t;
typedef bf16_t v16bf __attribute__((ext_vector_type(16)));
typedef bf16_t v8bf  __attribute__((ext_vector_type(8)));
typedef bf16_t v4bf  __attribute__((ext_vector_type(4)));
typedef float  v8f   __attribute__((ext_vector_type(8)));
typedef float  v4f   __attribute__((ext_vector_type(4)));

#define N_SPK   2048
#define M_UTT   16
#define P_DIM   512
#define ROWS    (N_SPK * M_UTT)     // 32768
#define BM      128
#define BN      128
#define BK      32
#define NKSTEP  (P_DIM / BK)        // 16
#define LDSS    40                  // padded LDS row stride (bf16); rows 80B, chunks 16B-aligned
#define ABUFB   (BM * LDSS * 2)     // bytes per LDS buffer

// round-to-nearest-even fp32 -> bf16
__device__ __forceinline__ bf16_t f2bf(float f) {
    unsigned u = __builtin_bit_cast(unsigned, f);
    unsigned r = (u + 0x7FFFu + ((u >> 16) & 1u)) >> 16;
    unsigned short s = (unsigned short)r;
    return __builtin_bit_cast(bf16_t, s);
}

// generic (__shared__) pointer -> 32-bit LDS byte offset (addrspace(3) ptrtoint)
typedef __attribute__((address_space(3))) const void lds_cvoid;
__device__ __forceinline__ unsigned lds_off(const void* p) {
    return (unsigned)(unsigned long long)(lds_cvoid*)p;
}

// 16-byte async DMA, GVS mode: LDS[ldsOff] = MEM[sbase + voff]  (ASYNCcnt)
__device__ __forceinline__ void async_cp16s(unsigned ldsOff, const void* sbase,
                                            unsigned voff) {
    asm volatile("global_load_async_to_lds_b128 %0, %1, %2"
                 :: "v"(ldsOff), "v"(voff), "s"(sbase) : "memory");
}
__device__ __forceinline__ void wait_async0() {
    asm volatile("s_wait_asynccnt 0x0" ::: "memory");
}

// ---------------------------------------------------------------------------
// Kernel 1: per-speaker centroid mean + bf16 stores + global sum-of-squares.
// inputs are read exactly once -> non-temporal loads. e_bf/cmean stores stay
// regular-temporal so the GEMM finds them in L2 (34MB << 192MB L2).
// ---------------------------------------------------------------------------
__global__ __launch_bounds__(256) void centroid_kernel(
    const float* __restrict__ inp, bf16_t* __restrict__ cmean_bf,
    bf16_t* __restrict__ e_bf, float* __restrict__ sumsq) {
    const int j = blockIdx.x;      // speaker 0..2047
    const int t = threadIdx.x;     // 0..255
    __shared__ float red[256];

    float local = 0.0f;
#pragma unroll
    for (int i = 0; i < P_DIM / 256; ++i) {
        const int p = t + i * 256;
        const size_t base = (size_t)j * M_UTT * P_DIM + p;
        float s = 0.0f;
#pragma unroll
        for (int mm = 0; mm < M_UTT; ++mm) {
            const float v = __builtin_nontemporal_load(inp + base + (size_t)mm * P_DIM);
            s += v;
            if (e_bf) e_bf[base + (size_t)mm * P_DIM] = f2bf(v);
        }
        const float cm = s * (1.0f / (float)M_UTT);
        cmean_bf[(size_t)j * P_DIM + p] = f2bf(cm);
        local += cm * cm;
    }

    red[t] = local;
    __syncthreads();
#pragma unroll
    for (int off = 128; off > 0; off >>= 1) {
        if (t < off) red[t] += red[t + off];
        __syncthreads();
    }
    if (t == 0) atomicAdd(sumsq, red[0]);
}

// ---------------------------------------------------------------------------
// WMMA compute for one 32x64 wave sub-tile from one staged K-chunk.
// ---------------------------------------------------------------------------
union AccU { v8f v; float f[8]; };
union FragU { v16bf v; v8bf h[2]; };

__device__ __forceinline__ void compute_step(
    const bf16_t* __restrict__ As, const bf16_t* __restrict__ Bs,
    AccU (&acc)[2][4], int waveRow, int waveCol, int m, int half) {
    FragU fa[2], fb[4];
#pragma unroll
    for (int rt = 0; rt < 2; ++rt) {
        const int ar = waveRow * 32 + rt * 16 + m;
        // 16-bit A layout: lanes 0-15 K{0..7,16..23}; lanes 16-31 K{8..15,24..31}
        fa[rt].h[0] = *(const v8bf*)&As[ar * LDSS + half * 8];
        fa[rt].h[1] = *(const v8bf*)&As[ar * LDSS + half * 8 + 16];
    }
#pragma unroll
    for (int ct = 0; ct < 4; ++ct) {
        const int br = waveCol * 64 + ct * 16 + m;
        // B layout: lanes 0-15 K{0..7,16..23}; lanes 16-31 K{8..15,24..31}
        fb[ct].h[0] = *(const v8bf*)&Bs[br * LDSS + half * 8];
        fb[ct].h[1] = *(const v8bf*)&Bs[br * LDSS + half * 8 + 16];
    }
#pragma unroll
    for (int rt = 0; rt < 2; ++rt)
#pragma unroll
        for (int ct = 0; ct < 4; ++ct)
            acc[rt][ct].v = __builtin_amdgcn_wmma_f32_16x16x32_bf16(
                false, fa[rt].v, false, fb[ct].v, (short)0, acc[rt][ct].v,
                false, false);
}

// ---------------------------------------------------------------------------
// Kernel 2: C[32768,2048] = A[32768,512] * B[2048,512]^T (bf16 WMMA, f32 acc)
// ABF16: A already bf16 -> SADDR async-to-LDS + double buffering.
// else : A fp32, convert while staging (sync loads, single buffer).
// ---------------------------------------------------------------------------
template <bool ABF16>
__global__ __launch_bounds__(256, 1) void wmma_gemm_kernel(
    const float* __restrict__ Af32, const bf16_t* __restrict__ Abf,
    const bf16_t* __restrict__ B,
    const float* __restrict__ wp, const float* __restrict__ bp,
    const float* __restrict__ sumsq, float* __restrict__ out) {
    __shared__ __align__(16) bf16_t As[2][BM * LDSS];
    __shared__ __align__(16) bf16_t Bs[2][BN * LDSS];

    const int tid  = threadIdx.x;
    const int lane = tid & 31;
    const int wave = tid >> 5;
    const int waveRow = wave & 3;       // 4 row groups -> 32 rows each
    const int waveCol = wave >> 2;      // 2 col groups -> 64 cols each
    const int m    = lane & 15;
    const int half = lane >> 4;

    const int R0 = blockIdx.y * BM;
    const int C0 = blockIdx.x * BN;

    AccU acc[2][4];
#pragma unroll
    for (int rt = 0; rt < 2; ++rt)
#pragma unroll
        for (int ct = 0; ct < 4; ++ct) acc[rt][ct].v = (v8f)0.0f;

    if constexpr (ABF16) {
        // ---- async double-buffered pipeline (GVS addressing) -------------
        // per thread per step: 2 A chunks + 2 B chunks, 16B each.
        // Precompute per-thread 32-bit byte offsets once; per stage just add
        // k0*2 (global) and buf*ABUFB (LDS).
        unsigned aoff[2], boff[2], ldsA[2], ldsB[2];
#pragma unroll
        for (int i = 0; i < 2; ++i) {
            const int c   = tid + 256 * i;          // 0..511
            const int row = c >> 2;
            const int col = (c & 3) * 8;
            aoff[i] = (unsigned)((R0 + row) * P_DIM + col) * 2u;
            boff[i] = (unsigned)((C0 + row) * P_DIM + col) * 2u;
            ldsA[i] = lds_off(&As[0][row * LDSS + col]);
            ldsB[i] = lds_off(&Bs[0][row * LDSS + col]);
        }
        auto stage = [&](int buf, int k0) {
            const unsigned kb   = (unsigned)k0 * 2u;
            const unsigned lbuf = (unsigned)buf * ABUFB;
#pragma unroll
            for (int i = 0; i < 2; ++i) {
                async_cp16s(ldsA[i] + lbuf, Abf, aoff[i] + kb);
                async_cp16s(ldsB[i] + lbuf, B,   boff[i] + kb);
            }
        };
        stage(0, 0);
        wait_async0();
        __syncthreads();
#pragma unroll
        for (int kk = 0; kk < NKSTEP; ++kk) {
            const int cur = kk & 1;
            if (kk + 1 < NKSTEP) stage(cur ^ 1, (kk + 1) * BK);
            compute_step(&As[cur][0], &Bs[cur][0], acc, waveRow, waveCol, m, half);
            wait_async0();       // my prefetch landed
            __syncthreads();     // everyone's prefetch landed / reads done
        }
    } else {
        // ---- fallback: fp32 A, convert while staging ---------------------
        for (int k0 = 0; k0 < P_DIM; k0 += BK) {
#pragma unroll
            for (int i = 0; i < 4; ++i) {
                const int c   = tid + 256 * i;       // 0..1023
                const int row = c >> 3;
                const int col = (c & 7) * 4;
                v4f f = *(const v4f*)(Af32 + (size_t)(R0 + row) * P_DIM + k0 + col);
                v4bf h;
                h[0] = f2bf(f[0]); h[1] = f2bf(f[1]);
                h[2] = f2bf(f[2]); h[3] = f2bf(f[3]);
                *(v4bf*)&As[0][row * LDSS + col] = h;
            }
#pragma unroll
            for (int i = 0; i < 2; ++i) {
                const int c   = tid + 256 * i;       // 0..511
                const int row = c >> 2;
                const int col = (c & 3) * 8;
                *(v8bf*)&Bs[0][row * LDSS + col] =
                    *(const v8bf*)(B + (size_t)(C0 + row) * P_DIM + k0 + col);
            }
            __syncthreads();
            compute_step(&As[0][0], &Bs[0][0], acc, waveRow, waveCol, m, half);
            __syncthreads();
        }
    }

    // ---- epilogue: alpha = |w| / sqrt(max(ss, eps)); clip; NT stores -----
    const float alpha = fabsf(wp[0]) / sqrtf(fmaxf(sumsq[0], 1e-6f));
    const float bias  = bp[0];
#pragma unroll
    for (int rt = 0; rt < 2; ++rt)
#pragma unroll
        for (int ct = 0; ct < 4; ++ct)
#pragma unroll
            for (int v = 0; v < 8; ++v) {
                const int row = R0 + waveRow * 32 + rt * 16 + (half ? v + 8 : v);
                const int col = C0 + waveCol * 64 + ct * 16 + m;
                const float s = alpha * acc[rt][ct].f[v] + bias;
                __builtin_nontemporal_store(fminf(fmaxf(s, 0.0f), 1.0f),
                                            out + (size_t)row * N_SPK + col);
            }
}

// ---------------------------------------------------------------------------
extern "C" void kernel_launch(void* const* d_in, const int* in_sizes, int n_in,
                              void* d_out, int out_size, void* d_ws, size_t ws_size,
                              hipStream_t stream) {
    const float* inputs = (const float*)d_in[0];   // [32768*512] fp32
    const float* w      = (const float*)d_in[1];   // [1]
    const float* b      = (const float*)d_in[2];   // [1]
    float*       out    = (float*)d_out;           // [32768*2048] fp32

    // ws layout: [0,4) sumsq | [256, 256+2MB) cmean_bf16 | then e_bf16 (32MB)
    const size_t cmean_bytes = (size_t)N_SPK * P_DIM * sizeof(bf16_t);
    const size_t ebf_bytes   = (size_t)ROWS * P_DIM * sizeof(bf16_t);
    float*  sumsq    = (float*)d_ws;
    bf16_t* cmean_bf = (bf16_t*)((char*)d_ws + 256);
    bf16_t* e_bf     = (bf16_t*)((char*)d_ws + 256 + cmean_bytes);
    const bool big   = ws_size >= 256 + cmean_bytes + ebf_bytes;

    hipMemsetAsync(sumsq, 0, sizeof(float), stream);

    centroid_kernel<<<N_SPK, 256, 0, stream>>>(inputs, cmean_bf,
                                               big ? e_bf : nullptr, sumsq);

    dim3 grid(N_SPK / BN, ROWS / BM);              // (16, 256)
    if (big)
        wmma_gemm_kernel<true><<<grid, 256, 0, stream>>>(
            inputs, e_bf, cmean_bf, w, b, sumsq, out);
    else
        wmma_gemm_kernel<false><<<grid, 256, 0, stream>>>(
            inputs, e_bf, cmean_bf, w, b, sumsq, out);
    (void)in_sizes; (void)n_in; (void)out_size; (void)ws_size;
}
